// GCNGenerator_57775900066051
// MI455X (gfx1250) — matrix operand order repaired and checked by reading, not compiled, not run on hardware
//
#include <hip/hip_runtime.h>
#include <hip/hip_bf16.h>

typedef __attribute__((ext_vector_type(16))) _Float16 v16h;
typedef __attribute__((ext_vector_type(8)))  _Float16 v8h;
typedef __attribute__((ext_vector_type(8)))  float    v8f;

#define NN   128          // nodes == features
#define PH   136          // f16 LDS row pitch (halfs) -> 272B rows, breaks bank conflicts
#define PF   132          // f32 LDS row pitch (floats) -> 528B rows
#define TPB  512          // 16 waves of 32

// ---- LDS layout offsets (bytes), all 16B aligned ----
#define OFF_ADJ  0
#define OFF_X    (NN * PF * 4)                 // 67584
#define OFF_Y    (OFF_X + NN * PH * 2)        // +34816
#define OFF_W    (OFF_Y + NN * PH * 2)
#define OFF_DINV (OFF_W + NN * PH * 2)
#define SMEM_BYTES (OFF_DINV + NN * 4)

// A-operand (16x32 f16 tile) from f16 row-major LDS, per CDNA5 WMMA VGPR layout:
// lanes 0-15 hold row M=lane, K chunks {0..7, 16..23}; lanes 16-31 hold M=lane-16, K {8..15, 24..31}
__device__ __forceinline__ v16h loadA16(const _Float16* A, int mt, int kt, int lane) {
    int m = mt * 16 + (lane & 15);
    const _Float16* p = A + m * PH + kt * 32 + ((lane & 16) ? 8 : 0);
    v8h lo = *(const v8h*)p;
    v8h hi = *(const v8h*)(p + 16);
    return __builtin_shufflevector(lo, hi, 0,1,2,3,4,5,6,7,8,9,10,11,12,13,14,15);
}

// Same A layout but source is f32 (adj): convert on the fly
__device__ __forceinline__ v16h loadA32(const float* A, int mt, int kt, int lane) {
    int m = mt * 16 + (lane & 15);
    const float* p = A + m * PF + kt * 32 + ((lane & 16) ? 8 : 0);
    v16h a;
#pragma unroll
    for (int j = 0; j < 8; ++j) {
        a[j]     = (_Float16)p[j];
        a[j + 8] = (_Float16)p[j + 16];
    }
    return a;
}

// B-operand (32x16 f16 tile): lane L holds col N=L%16; lanes 0-15 K {0..7, 16..23}, lanes 16-31 K {8..15, 24..31}
__device__ __forceinline__ v16h loadB(const _Float16* Bm, int kt, int nt, int lane) {
    const _Float16* p = Bm + (kt * 32 + ((lane & 16) ? 8 : 0)) * PH + nt * 16 + (lane & 15);
    v16h b;
#pragma unroll
    for (int j = 0; j < 8; ++j) {
        b[j]     = p[j * PH];
        b[j + 8] = p[(j + 16) * PH];
    }
    return b;
}

// D (16x16 f32) -> f16 row-major store. Lane L<16 holds col N=L, rows M=0..7 in acc[0..7];
// lane>=16 holds col N=L-16, rows M=8..15.
template <bool RELU>
__device__ __forceinline__ void storeD(_Float16* D, int mt, int nt, int lane, v8f acc) {
    int n  = nt * 16 + (lane & 15);
    int mb = mt * 16 + ((lane & 16) ? 8 : 0);
#pragma unroll
    for (int r = 0; r < 8; ++r) {
        float v = acc[r];
        if (RELU) v = v > 0.f ? v : 0.f;
        D[(mb + r) * PH + n] = (_Float16)v;
    }
}

// D = [relu](A_f16 @ B)   128x128x128, 16 waves x 4 tiles, K split into 4 WMMAs of 32
template <bool RELU>
__device__ __forceinline__ void mm_f16(const _Float16* A, const _Float16* Bm,
                                       _Float16* D, int wave, int lane) {
#pragma unroll
    for (int t = 0; t < 4; ++t) {
        int tile = wave * 4 + t;
        int mt = tile >> 3, nt = tile & 7;
        v8f acc = {};
#pragma unroll
        for (int kt = 0; kt < 4; ++kt) {
            v16h a = loadA16(A, mt, kt, lane);
            v16h b = loadB(Bm, kt, nt, lane);
            acc = __builtin_amdgcn_wmma_f32_16x16x32_f16(false, a, false, b,
                                                         (short)0, acc, false, false);
        }
        storeD<RELU>(D, mt, nt, lane, acc);
    }
}

// D = A_f32 @ B  (A = adj, converted per-fetch)
__device__ __forceinline__ void mm_f32A(const float* A, const _Float16* Bm,
                                        _Float16* D, int wave, int lane) {
#pragma unroll
    for (int t = 0; t < 4; ++t) {
        int tile = wave * 4 + t;
        int mt = tile >> 3, nt = tile & 7;
        v8f acc = {};
#pragma unroll
        for (int kt = 0; kt < 4; ++kt) {
            v16h a = loadA32(A, mt, kt, lane);
            v16h b = loadB(Bm, kt, nt, lane);
            acc = __builtin_amdgcn_wmma_f32_16x16x32_f16(false, a, false, b,
                                                         (short)0, acc, false, false);
        }
        storeD<false>(D, mt, nt, lane, acc);
    }
}

extern "C" __global__ __launch_bounds__(TPB)
void gcn_generator_kernel(const float* __restrict__ gx,
                          const float* __restrict__ gw,
                          float* __restrict__ gout) {
    extern __shared__ char smem[];
    float*    s_adj  = (float*)(smem + OFF_ADJ);
    _Float16* s_x    = (_Float16*)(smem + OFF_X);
    _Float16* s_y    = (_Float16*)(smem + OFF_Y);
    _Float16* s_w    = (_Float16*)(smem + OFF_W);
    float*    s_dinv = (float*)(smem + OFF_DINV);

    const int tid  = threadIdx.x;
    const int lane = tid & 31;
    const int wave = tid >> 5;
    const float* xb   = gx   + (size_t)blockIdx.x * NN * NN;
    float*       outb = gout + (size_t)blockIdx.x * NN * NN;

    // ---- init: W,x -> f16 LDS; adj = I; out = I
    for (int e = tid; e < NN * NN; e += TPB) {
        int r = e >> 7, c = e & 127;
        s_w[r * PH + c] = (_Float16)gw[e];
        s_y[r * PH + c] = (_Float16)xb[e];    // original x staged in y-buffer
        float id = (r == c) ? 1.0f : 0.0f;
        s_adj[r * PF + c] = id;
        outb[e] = id;
    }
    __syncthreads();

    // x = relu(x @ W)
    mm_f16<true>(s_y, s_w, s_x, wave, lane);
    __syncthreads();

    for (int i = 1; i < NN; ++i) {
        // ---- prob[j] = x[j] . x[i]; masked writes to out & adj (row i / col i, j < i)
        if (tid < NN) {
            const _Float16* xr = s_x + tid * PH;
            const _Float16* xi = s_x + i * PH;
            float acc = 0.f;
#pragma unroll 4
            for (int kk = 0; kk < NN; kk += 8) {
                v8h a = *(const v8h*)(xr + kk);
                v8h c = *(const v8h*)(xi + kk);
#pragma unroll
                for (int j = 0; j < 8; ++j) acc += (float)a[j] * (float)c[j];
            }
            if (tid < i) {
                outb[i * NN + tid]  = acc;
                outb[tid * NN + i]  = acc;
                s_adj[i * PF + tid] = acc;
                s_adj[tid * PF + i] = acc;
            }
        }
        __syncthreads();

        // ---- deg -> dinv = deg^-0.5
        if (tid < NN) {
            const float* ar = s_adj + tid * PF;
            float s = 0.f;
#pragma unroll 8
            for (int k = 0; k < NN; ++k) s += ar[k];
            s_dinv[tid] = rsqrtf(s);
        }
        __syncthreads();

        // ---- adj = dinv[r] * adj * dinv[c]
        {
            int r  = tid >> 2;
            int c0 = (tid & 3) * 32;
            float dr = s_dinv[r];
            float* ar = s_adj + r * PF + c0;
#pragma unroll 8
            for (int c = 0; c < 32; ++c) ar[c] = ar[c] * dr * s_dinv[c0 + c];
        }
        __syncthreads();

        // ---- y = adj @ x
        mm_f32A(s_adj, s_x, s_y, wave, lane);
        __syncthreads();

        // ---- x = relu(y @ W)
        mm_f16<true>(s_y, s_w, s_x, wave, lane);
        __syncthreads();
    }
}

extern "C" void kernel_launch(void* const* d_in, const int* in_sizes, int n_in,
                              void* d_out, int out_size, void* d_ws, size_t ws_size,
                              hipStream_t stream) {
    const float* x = (const float*)d_in[0];
    const float* W = (const float*)d_in[1];
    float* out = (float*)d_out;
    int batches = in_sizes[0] / (NN * NN);   // 32
    gcn_generator_kernel<<<batches, TPB, SMEM_BYTES, stream>>>(x, W, out);
}
